// ScannedMemoroid_20607253086805
// MI455X (gfx1250) — matrix operand order repaired and checked by reading, not compiled, not run on hardware
//
#include <hip/hip_runtime.h>
#include <hip/hip_bf16.h>
#include <stdint.h>

// Problem sizes (fixed by the reference)
#define TT 4096
#define BB 16
#define HH 256
#define DMODEL 256
#define MROWS (TT*BB)          // 65536 flattened (t,b) rows
#define NCHUNK 32
#define CLEN (TT/NCHUNK)       // 128 timesteps per chunk

typedef __attribute__((ext_vector_type(16))) __bf16 v16bf;
typedef __attribute__((ext_vector_type(8)))  float  v8f;

// ---------------- workspace layout (bytes) ----------------
static constexpr size_t WS_LAMRE = 0;
static constexpr size_t WS_LAMIM = 1024;
static constexpr size_t WS_GAMMA = 2048;
static constexpr size_t WS_BG    = 4096;                                  // bf16 [512][256]
static constexpr size_t WS_CC    = WS_BG + (size_t)512*256*2;             // bf16 [256][512]
static constexpr size_t WS_BURE  = WS_CC + (size_t)256*512*2;             // bf16 [65536][256]
static constexpr size_t WS_BUIM  = WS_BURE + (size_t)MROWS*HH*2;          // bf16 [65536][256]
static constexpr size_t WS_HCAT  = WS_BUIM + (size_t)MROWS*HH*2;          // bf16 [65536][512]
static constexpr size_t WS_AGG   = WS_HCAT + (size_t)MROWS*512*2;
static constexpr size_t AGG_N    = (size_t)NCHUNK*BB*HH;                  // 131072
static constexpr size_t WS_AGG_ARE = WS_AGG;
static constexpr size_t WS_AGG_AIM = WS_AGG + AGG_N*4;
static constexpr size_t WS_AGG_BRE = WS_AGG + AGG_N*8;
static constexpr size_t WS_AGG_BIM = WS_AGG + AGG_N*12;
static constexpr size_t WS_AGG_D   = WS_AGG + AGG_N*16;
static constexpr size_t WS_CARRE   = WS_AGG + AGG_N*20;
static constexpr size_t WS_CARIM   = WS_CARRE + AGG_N*4;

// ---------------- helpers ----------------
// Plain RNE bit-twiddle (non-hot paths)
__device__ __forceinline__ uint16_t f2bf(float f) {
  union { float f; uint32_t u; } v; v.f = f;
  return (uint16_t)((v.u + 0x7FFFu + ((v.u >> 16) & 1u)) >> 16);   // RNE
}
__device__ __forceinline__ float bf2f(uint16_t u) {
  union { uint32_t u; float f; } v; v.u = ((uint32_t)u) << 16;
  return v.f;
}
// CDNA5 VOP3P mix ops: bf16(fma(x,1.0,0)) into low/high half — 2 VALU per pair.
__device__ __forceinline__ uint32_t pk_bf16(float lo, float hi) {
  uint32_t d;
  asm("v_fma_mixlo_bf16 %0, %1, 1.0, 0\n\t"
      "v_fma_mixhi_bf16 %0, %2, 1.0, 0"
      : "=&v"(d) : "v"(lo), "v"(hi));
  return d;
}
__device__ __forceinline__ uint16_t f2bf_fast(float f) {
  uint32_t d;
  asm("v_fma_mixlo_bf16 %0, %1, 1.0, 0" : "=v"(d) : "v"(f));
  return (uint16_t)d;
}

// ---------------- kernel 0a: Lambda / gamma scalars ----------------
__global__ void k_scalars(const float* __restrict__ nu_log, const float* __restrict__ theta_log,
                          float* __restrict__ lamre, float* __restrict__ lamim,
                          float* __restrict__ gamma) {
  int h = threadIdx.x;
  if (h < HH) {
    float nu = __expf(nu_log[h]);
    float th = __expf(theta_log[h]);
    float r  = __expf(-nu);                 // |Lambda|
    lamre[h] = r * __cosf(th);
    lamim[h] = r * __sinf(th);
    gamma[h] = sqrtf(fmaxf(1.0f - r * r, 1e-8f));
  }
}

// ---------------- kernel 0b: pack weights to bf16 ----------------
__global__ void k_pack(const float* __restrict__ Bre, const float* __restrict__ Bim,
                       const float* __restrict__ Cre, const float* __restrict__ Cim,
                       const float* __restrict__ gamma,
                       uint16_t* __restrict__ bg, uint16_t* __restrict__ cc) {
  int idx = blockIdx.x * blockDim.x + threadIdx.x;   // 0..131071
  {
    int n = idx >> 8, d = idx & 255;
    int h = n & 255;
    float w = ((n < 256) ? Bre[h * DMODEL + d] : Bim[h * DMODEL + d]) * gamma[h];
    bg[idx] = f2bf(w);
  }
  {
    int d = idx >> 9, k = idx & 511;
    float c = (k < 256) ? Cre[d * HH + k] : -Cim[d * HH + (k - 256)];
    cc[idx] = f2bf(c);
  }
}

// ---------------- kernel 1: Bu = x @ Bgcat^T (bf16 WMMA, f32 accum) ----------------
// grid: 8192 blocks x 128 threads (4 waves). Wave tile: 16M x 64N, K=256 in 8 steps.
// Explicit ping-pong double buffering: next K-step's loads are in flight while
// current step's WMMAs execute (breaks the per-MMA s_wait_loadcnt 0 drain).
__global__ void __launch_bounds__(128) k_gemm_bu(const float* __restrict__ x,
                                                 const uint16_t* __restrict__ bg,
                                                 uint16_t* __restrict__ bure,
                                                 uint16_t* __restrict__ buim) {
  const int lane  = threadIdx.x & 31;
  const int wave  = threadIdx.x >> 5;
  const int mtile = blockIdx.x >> 1;
  const int nhalf = blockIdx.x & 1;
  const int Mbase = mtile * 16;
  const int Nbase = nhalf * 256 + wave * 64;
  const int row   = lane & 15;
  const int hi    = lane >> 4;

  const float* xrow = x + (size_t)(Mbase + row) * DMODEL;

  v8f acc[4];
#pragma unroll
  for (int j = 0; j < 4; ++j) acc[j] = (v8f){0.f,0.f,0.f,0.f,0.f,0.f,0.f,0.f};

  union ABu { v16bf v; uint32_t u[8]; uint4 q[2]; };

  float4 ar[2][4];
  ABu    Bf[2][4];

  auto load_step = [&](int k0, int buf) {
    const int ka = k0 + hi * 8;
    ar[buf][0] = *(const float4*)(xrow + ka);
    ar[buf][1] = *(const float4*)(xrow + ka + 4);
    ar[buf][2] = *(const float4*)(xrow + ka + 16);
    ar[buf][3] = *(const float4*)(xrow + ka + 20);
#pragma unroll
    for (int j = 0; j < 4; ++j) {
      const int n = Nbase + j * 16 + row;
      const uint8_t* bp = (const uint8_t*)bg + ((size_t)n * DMODEL + k0 + hi * 8) * 2;
      Bf[buf][j].q[0] = *(const uint4*)(bp);
      Bf[buf][j].q[1] = *(const uint4*)(bp + 32);
    }
  };

  load_step(0, 0);
#pragma unroll
  for (int s = 0; s < 8; ++s) {
    const int cur = s & 1;
    if (s < 7) load_step((s + 1) * 32, cur ^ 1);

    // Pack A (lane<16: K {base..base+7, base+16..base+23}; lane>=16 shifted by 8)
    ABu A;
    A.u[0] = pk_bf16(ar[cur][0].x, ar[cur][0].y); A.u[1] = pk_bf16(ar[cur][0].z, ar[cur][0].w);
    A.u[2] = pk_bf16(ar[cur][1].x, ar[cur][1].y); A.u[3] = pk_bf16(ar[cur][1].z, ar[cur][1].w);
    A.u[4] = pk_bf16(ar[cur][2].x, ar[cur][2].y); A.u[5] = pk_bf16(ar[cur][2].z, ar[cur][2].w);
    A.u[6] = pk_bf16(ar[cur][3].x, ar[cur][3].y); A.u[7] = pk_bf16(ar[cur][3].z, ar[cur][3].w);

#pragma unroll
    for (int j = 0; j < 4; ++j)
      acc[j] = __builtin_amdgcn_wmma_f32_16x16x32_bf16(false, A.v, false, Bf[cur][j].v,
                                                       (short)0, acc[j], false, false);
  }

  // D layout: VGPR r -> row r (lanes 0-15) / r+8 (lanes 16-31), col = lane%16
#pragma unroll
  for (int j = 0; j < 4; ++j) {
    const int n  = Nbase + j * 16 + row;
    const int mb = Mbase + hi * 8;
    uint16_t* dst = (n < HH) ? (bure + n) : (buim + (n - HH));
#pragma unroll
    for (int r = 0; r < 8; ++r)
      dst[(size_t)(mb + r) * HH] = f2bf_fast(acc[j][r]);
  }
}

// ---------------- kernel 2a: per-chunk scan aggregates ----------------
__global__ void __launch_bounds__(256) k_scan_agg(const uint16_t* __restrict__ bure,
                                                  const uint16_t* __restrict__ buim,
                                                  const unsigned char* __restrict__ starts,
                                                  const float* __restrict__ lamre,
                                                  const float* __restrict__ lamim,
                                                  float* __restrict__ are, float* __restrict__ aim,
                                                  float* __restrict__ bre, float* __restrict__ bim,
                                                  float* __restrict__ dfl) {
  const int h = threadIdx.x;
  const int b = blockIdx.x & (BB - 1);
  const int c = blockIdx.x >> 4;
  const float lre = lamre[h], lim = lamim[h];

  float Are = 1.f, Aim = 0.f, Sre = 0.f, Sim = 0.f, dflag = 0.f;
  const int t0 = c * CLEN;
  for (int i = 0; i < CLEN; ++i) {
    const size_t m = (size_t)(t0 + i) * BB + b;
    const float bu_re = bf2f(bure[m * HH + h]);
    const float bu_im = bf2f(buim[m * HH + h]);
    if (starts[m]) {
      Are = lre; Aim = lim; Sre = bu_re; Sim = bu_im; dflag = 1.f;
    } else {
      const float nAre = lre * Are - lim * Aim;
      const float nAim = lre * Aim + lim * Are;
      Are = nAre; Aim = nAim;
      const float nSre = lre * Sre - lim * Sim + bu_re;
      const float nSim = lre * Sim + lim * Sre + bu_im;
      Sre = nSre; Sim = nSim;
    }
  }
  const size_t o = (size_t)blockIdx.x * HH + h;   // [(c*16+b)*256 + h]
  are[o] = Are; aim[o] = Aim; bre[o] = Sre; bim[o] = Sim; dfl[o] = dflag;
}

// ---------------- kernel 2b: scan over chunk aggregates -> carry-in per chunk ----------------
__global__ void __launch_bounds__(256) k_scan_carry(const float* __restrict__ are, const float* __restrict__ aim,
                                                    const float* __restrict__ bre, const float* __restrict__ bim,
                                                    const float* __restrict__ dfl,
                                                    const float* __restrict__ state_re,
                                                    const float* __restrict__ state_im,
                                                    float* __restrict__ cre, float* __restrict__ cim) {
  const int h = threadIdx.x;
  const int b = blockIdx.x;
  float Sre = state_re[b * HH + h];
  float Sim = state_im[b * HH + h];
  for (int c = 0; c < NCHUNK; ++c) {
    const size_t o = ((size_t)c * BB + b) * HH + h;
    cre[o] = Sre; cim[o] = Sim;
    const float Are = are[o], Aim = aim[o], Bre_ = bre[o], Bim_ = bim[o];
    if (dfl[o] != 0.f) { Sre = Bre_; Sim = Bim_; }
    else {
      const float nre = Are * Sre - Aim * Sim + Bre_;
      const float nim = Are * Sim + Aim * Sre + Bim_;
      Sre = nre; Sim = nim;
    }
  }
}

// ---------------- kernel 2c: apply carries, emit hcat bf16 [m][h_re||h_im] ----------------
__global__ void __launch_bounds__(256) k_scan_apply(const uint16_t* __restrict__ bure,
                                                    const uint16_t* __restrict__ buim,
                                                    const unsigned char* __restrict__ starts,
                                                    const float* __restrict__ lamre,
                                                    const float* __restrict__ lamim,
                                                    const float* __restrict__ cre,
                                                    const float* __restrict__ cim,
                                                    uint16_t* __restrict__ hcat) {
  const int h = threadIdx.x;
  const int b = blockIdx.x & (BB - 1);
  const int c = blockIdx.x >> 4;
  const float lre = lamre[h], lim = lamim[h];
  const size_t o = ((size_t)c * BB + b) * HH + h;
  float Sre = cre[o], Sim = cim[o];
  const int t0 = c * CLEN;
  for (int i = 0; i < CLEN; ++i) {
    const size_t m = (size_t)(t0 + i) * BB + b;
    const float bu_re = bf2f(bure[m * HH + h]);
    const float bu_im = bf2f(buim[m * HH + h]);
    if (starts[m]) { Sre = bu_re; Sim = bu_im; }
    else {
      const float nre = lre * Sre - lim * Sim + bu_re;
      const float nim = lre * Sim + lim * Sre + bu_im;
      Sre = nre; Sim = nim;
    }
    hcat[m * 512 + h]       = f2bf_fast(Sre);
    hcat[m * 512 + 256 + h] = f2bf_fast(Sim);
  }
}

// ---------------- kernel 3: out = hcat @ Ccat^T + D*x (bf16 WMMA) ----------------
// grid: 4096 blocks x 128 threads. Wave tile 16M x 64N, K=512 in 16 steps, ping-pong.
__global__ void __launch_bounds__(128) k_gemm_out(const uint16_t* __restrict__ hcat,
                                                  const uint16_t* __restrict__ cc,
                                                  const float* __restrict__ x,
                                                  const float* __restrict__ Dd,
                                                  float* __restrict__ out) {
  const int lane  = threadIdx.x & 31;
  const int wave  = threadIdx.x >> 5;
  const int Mbase = blockIdx.x * 16;
  const int Nbase = wave * 64;
  const int row   = lane & 15;
  const int hi    = lane >> 4;

  const uint8_t* arow = (const uint8_t*)hcat + (size_t)(Mbase + row) * 512 * 2;

  v8f acc[4];
#pragma unroll
  for (int j = 0; j < 4; ++j) acc[j] = (v8f){0.f,0.f,0.f,0.f,0.f,0.f,0.f,0.f};

  union ABu { v16bf v; uint4 q[2]; };

  ABu Aq[2];
  ABu Bf[2][4];

  auto load_step = [&](int k0, int buf) {
    const uint8_t* ap = arow + (size_t)(k0 + hi * 8) * 2;
    Aq[buf].q[0] = *(const uint4*)(ap);        // K run {base..base+7}
    Aq[buf].q[1] = *(const uint4*)(ap + 32);   // K run {base+16..base+23}
#pragma unroll
    for (int j = 0; j < 4; ++j) {
      const int n = Nbase + j * 16 + row;
      const uint8_t* bp = (const uint8_t*)cc + ((size_t)n * 512 + k0 + hi * 8) * 2;
      Bf[buf][j].q[0] = *(const uint4*)(bp);
      Bf[buf][j].q[1] = *(const uint4*)(bp + 32);
    }
  };

  load_step(0, 0);
#pragma unroll
  for (int s = 0; s < 16; ++s) {
    const int cur = s & 1;
    if (s < 15) load_step((s + 1) * 32, cur ^ 1);
#pragma unroll
    for (int j = 0; j < 4; ++j)
      acc[j] = __builtin_amdgcn_wmma_f32_16x16x32_bf16(false, Aq[cur].v, false, Bf[cur][j].v,
                                                       (short)0, acc[j], false, false);
  }

#pragma unroll
  for (int j = 0; j < 4; ++j) {
    const int n = Nbase + j * 16 + row;
    const float dn = Dd[n];
#pragma unroll
    for (int r = 0; r < 8; ++r) {
      const int m = Mbase + hi * 8 + r;
      const size_t off = (size_t)m * DMODEL + n;
      out[off] = acc[j][r] + dn * x[off];
    }
  }
}

// ---------------- launch ----------------
extern "C" void kernel_launch(void* const* d_in, const int* in_sizes, int n_in,
                              void* d_out, int out_size, void* d_ws, size_t ws_size,
                              hipStream_t stream) {
  const float* x          = (const float*)d_in[0];
  const unsigned char* st = (const unsigned char*)d_in[1];   // bool array
  const float* state_re   = (const float*)d_in[2];
  const float* state_im   = (const float*)d_in[3];
  const float* nu_log     = (const float*)d_in[4];
  const float* theta_log  = (const float*)d_in[5];
  const float* B_re       = (const float*)d_in[6];
  const float* B_im       = (const float*)d_in[7];
  const float* C_re       = (const float*)d_in[8];
  const float* C_im       = (const float*)d_in[9];
  const float* Dvec       = (const float*)d_in[10];
  float* out = (float*)d_out;

  char* ws = (char*)d_ws;
  float*    lamre = (float*)(ws + WS_LAMRE);
  float*    lamim = (float*)(ws + WS_LAMIM);
  float*    gamma = (float*)(ws + WS_GAMMA);
  uint16_t* bg    = (uint16_t*)(ws + WS_BG);
  uint16_t* cc    = (uint16_t*)(ws + WS_CC);
  uint16_t* bure  = (uint16_t*)(ws + WS_BURE);
  uint16_t* buim  = (uint16_t*)(ws + WS_BUIM);
  uint16_t* hcat  = (uint16_t*)(ws + WS_HCAT);
  float*    are   = (float*)(ws + WS_AGG_ARE);
  float*    aim   = (float*)(ws + WS_AGG_AIM);
  float*    bre   = (float*)(ws + WS_AGG_BRE);
  float*    bim   = (float*)(ws + WS_AGG_BIM);
  float*    dfl   = (float*)(ws + WS_AGG_D);
  float*    cre   = (float*)(ws + WS_CARRE);
  float*    cim   = (float*)(ws + WS_CARIM);

  k_scalars<<<1, 256, 0, stream>>>(nu_log, theta_log, lamre, lamim, gamma);
  k_pack<<<512, 256, 0, stream>>>(B_re, B_im, C_re, C_im, gamma, bg, cc);
  k_gemm_bu<<<MROWS / 16 * 2, 128, 0, stream>>>(x, bg, bure, buim);
  k_scan_agg<<<NCHUNK * BB, 256, 0, stream>>>(bure, buim, st, lamre, lamim, are, aim, bre, bim, dfl);
  k_scan_carry<<<BB, 256, 0, stream>>>(are, aim, bre, bim, dfl, state_re, state_im, cre, cim);
  k_scan_apply<<<NCHUNK * BB, 256, 0, stream>>>(bure, buim, st, lamre, lamim, cre, cim, hcat);
  k_gemm_out<<<MROWS / 16, 128, 0, stream>>>(hcat, cc, x, Dvec, out);
}
